// EchoChamber_11965778886781
// MI455X (gfx1250) — matrix operand order repaired and checked by reading, not compiled, not run on hardware
//
#include <hip/hip_runtime.h>
#include <math.h>

typedef __attribute__((ext_vector_type(16))) _Float16 v16h;
typedef __attribute__((ext_vector_type(8)))  float    v8f;

#define LUT_N           4096
#define D_DIM           1024
#define H_DIM           4
#define WAVES_PER_BLOCK 8
#define ROWS_PER_BLOCK  (WAVES_PER_BLOCK * 16)

// Faithful reproduction of reference lookup_sin_cos (LUT linear interpolation).
__device__ __forceinline__ void lut_sincos(const float2* __restrict__ lut,
                                           float th, float& s, float& c) {
  float pos = th * 651.89864690440329f;           // LUT_N / (2*pi)
  float fl  = floorf(pos);
  float fr  = pos - fl;
  int i0 = ((int)fl) & (LUT_N - 1);               // two's-complement & == positive mod
  int i1 = (i0 + 1) & (LUT_N - 1);
  float2 e0 = lut[i0];
  float2 e1 = lut[i1];
  s = fmaf(fr, e1.x - e0.x, e0.x);                // s0*(1-f)+s1*f
  c = fmaf(fr, e1.y - e0.y, e0.y);
}

__global__ __launch_bounds__(256) void echo_fused(
    const float* __restrict__ x_real, const float* __restrict__ x_imag,
    const float* __restrict__ t_in,
    const float* __restrict__ trig_r, const float* __restrict__ trig_i,
    const float* __restrict__ w_query, const float* __restrict__ b_query,
    const float* __restrict__ w_out,  const float* __restrict__ b_out,
    const float* __restrict__ beta,
    float* __restrict__ out_real, float* __restrict__ out_imag, int B)
{
  __shared__ float2 sLut[LUT_N];                  // 32 KB: interleaved (sin, cos)
  __shared__ float  sInvWl[D_DIM];                // 1/(1+|w_query|)
  __shared__ float  sBq[D_DIM];
  __shared__ float  sTrS[D_DIM];                  // trigger_real / H
  __shared__ float  sTiS[D_DIM];                  // trigger_imag / H
  __shared__ float  sWe[D_DIM];                   // 1/(1+|w_out|)
  __shared__ float  sWs[D_DIM];                   // write_scale
  __shared__ float  sBo[D_DIM];
  __shared__ float  sTphi[ROWS_PER_BLOCK];
  __shared__ float  sTot[WAVES_PER_BLOCK][2][16];
  __shared__ float  sMag[WAVES_PER_BLOCK][16];

  const int tid = threadIdx.x;

  // ---- Stage LUT and per-D parameters into LDS ----
  for (int i = tid; i < LUT_N; i += 256) {
    float ang = (float)i * 0.00153398078788564123f;   // 2*pi/LUT_N
    sLut[i] = make_float2(sinf(ang), cosf(ang));
  }
  for (int d = tid; d < D_DIM; d += 256) {
    sInvWl[d] = 1.0f / (1.0f + fabsf(w_query[d]));
    sBq[d]    = b_query[d];
    sTrS[d]   = trig_r[d] * (1.0f / (float)H_DIM);
    sTiS[d]   = trig_i[d] * (1.0f / (float)H_DIM);
    float we  = 1.0f / (1.0f + fabsf(w_out[d]));
    float be  = 1.0f / (1.0f + fabsf(beta[d]));
    float dec = fminf(expf(-be * we), 0.9999f);
    sWe[d] = we;
    sWs[d] = we * (1.0f - dec);
    sBo[d] = b_out[d];
  }
  const int wgRow0 = blockIdx.x * ROWS_PER_BLOCK;
  if (tid < ROWS_PER_BLOCK) {
    int r = wgRow0 + tid;
    if (r >= B) r = B - 1;
    sTphi[tid] = t_in[r] * 1.61803398874989485f;      // t * PHI
  }
  __syncthreads();

  const int w    = tid >> 5;        // wave (wave32)
  const int lane = tid & 31;
  const int n    = lane & 15;       // A: row within tile; B/C: column
  const int g    = lane >> 4;       // lane group

  // ---------------- Phase 1: WMMA reduction -> tot_r/tot_i ----------------
  // A (16x32 f16): row m=n; K-interleaved (q_r,q_i). Lane-group K mapping:
  //   h<8 : K = 8g+h   -> d = 16c + 4g + h/2
  //   h>=8: K = 8+8g+h -> d = 16c + 8 + 4g + (h-8)/2
  // B (32x16 f16): col n; same K mapping as A (h<8: d = 16c + 4g + h/2; h>=8: d = 16c + 8 + 4g + (h-8)/2).
  //   col0 = (Tr', Ti') interleave  -> tot_r ;  col1 = (Ti', -Tr') -> tot_i
  const int rowLocal = w * 16 + n;
  int rowA = wgRow0 + rowLocal;
  if (rowA >= B) rowA = B - 1;                        // clamp loads; stores guarded
  const float tphiA = sTphi[rowLocal];

  const float4* xr4A = (const float4*)(x_real + (size_t)rowA * D_DIM);
  const float4* xi4A = (const float4*)(x_imag + (size_t)rowA * D_DIM);

  v8f acc = {};

  for (int c = 0; c < D_DIM / 16; ++c) {
    const int dA0 = 16 * c + 4 * g;                   // first quad of d-values
    const int dB0 = dA0 + 8;                          // second quad

    float4 xrA = xr4A[dA0 >> 2];
    float4 xiA = xi4A[dA0 >> 2];
    float4 xrB = xr4A[dB0 >> 2];
    float4 xiB = xi4A[dB0 >> 2];
    float4 iwA = *(const float4*)&sInvWl[dA0];
    float4 iwB = *(const float4*)&sInvWl[dB0];
    float4 bqA = *(const float4*)&sBq[dA0];
    float4 bqB = *(const float4*)&sBq[dB0];

    v16h a;
#pragma unroll
    for (int q = 0; q < 4; ++q) {
      {
        float off = reinterpret_cast<const float*>(&bqA)[q] + tphiA;
        float iw  = reinterpret_cast<const float*>(&iwA)[q];
        float thr = fmaf(reinterpret_cast<const float*>(&xrA)[q], iw, off);
        float thi = fmaf(reinterpret_cast<const float*>(&xiA)[q], iw, off);
        float sr, cr, si, ci;
        lut_sincos(sLut, thr, sr, cr);
        lut_sincos(sLut, thi, si, ci);
        a[2 * q]     = (_Float16)(cr * ci - sr * si); // q_r
        a[2 * q + 1] = (_Float16)(cr * si + sr * ci); // q_i
      }
      {
        float off = reinterpret_cast<const float*>(&bqB)[q] + tphiA;
        float iw  = reinterpret_cast<const float*>(&iwB)[q];
        float thr = fmaf(reinterpret_cast<const float*>(&xrB)[q], iw, off);
        float thi = fmaf(reinterpret_cast<const float*>(&xiB)[q], iw, off);
        float sr, cr, si, ci;
        lut_sincos(sLut, thr, sr, cr);
        lut_sincos(sLut, thi, si, ci);
        a[8 + 2 * q] = (_Float16)(cr * ci - sr * si);
        a[9 + 2 * q] = (_Float16)(cr * si + sr * ci);
      }
    }

    const int dT0 = 16 * c + 4 * g;
    float4 trL = *(const float4*)&sTrS[dT0];
    float4 trH = *(const float4*)&sTrS[dT0 + 8];
    float4 tiL = *(const float4*)&sTiS[dT0];
    float4 tiH = *(const float4*)&sTiS[dT0 + 8];

    v16h bf;
#pragma unroll
    for (int k = 0; k < 8; ++k) {
      float tr = (k < 4) ? reinterpret_cast<const float*>(&trL)[k]
                         : reinterpret_cast<const float*>(&trH)[k - 4];
      float ti = (k < 4) ? reinterpret_cast<const float*>(&tiL)[k]
                         : reinterpret_cast<const float*>(&tiH)[k - 4];
      float ev = (n == 0) ? tr : ((n == 1) ? ti  : 0.0f);
      float od = (n == 0) ? ti : ((n == 1) ? -tr : 0.0f);
      bf[2 * k]     = (_Float16)ev;
      bf[2 * k + 1] = (_Float16)od;
    }

    acc = __builtin_amdgcn_wmma_f32_16x16x32_f16(
        /*neg_a=*/false, a, /*neg_b=*/false, bf,
        /*c_mod=*/(short)0, acc, /*reuse_a=*/false, /*reuse_b=*/false);
  }

  // C layout: VGPR r, lanes 0-15 -> M=r, lanes 16-31 -> M=8+r; N = lane%16.
  if (n < 2) {
#pragma unroll
    for (int r = 0; r < 8; ++r) sTot[w][n][g * 8 + r] = acc[r];
  }
  __syncthreads();
  if (lane < 16) {
    float tr = sTot[w][0][n];
    float ti = sTot[w][1][n];
    sMag[w][n] = sqrtf(fmaf(tr, tr, fmaf(ti, ti, 1e-8f)));
  }
  __syncthreads();

  // ---------------- Phase 2: streaming output (x re-read hits L2) ----------
  for (int m = 0; m < 16; ++m) {
    const int row = wgRow0 + w * 16 + m;
    if (row >= B) break;                              // uniform within wave
    const float mag = sMag[w][m];
    const float tp  = sTphi[w * 16 + m];
    const float4* xr4 = (const float4*)(x_real + (size_t)row * D_DIM);
    const float4* xi4 = (const float4*)(x_imag + (size_t)row * D_DIM);
    float4* po_r = (float4*)(out_real + (size_t)row * D_DIM);
    float4* po_i = (float4*)(out_imag + (size_t)row * D_DIM);
#pragma unroll 1
    for (int it = 0; it < D_DIM / 128; ++it) {
      const int d4 = it * 32 + lane;                  // float4 index (coalesced)
      const int d  = d4 * 4;
      float4 xr = xr4[d4];
      float4 xi = xi4[d4];
      float4 ws = *(const float4*)&sWs[d];
      float4 we = *(const float4*)&sWe[d];
      float4 bo = *(const float4*)&sBo[d];
      float4 oR, oI;
#pragma unroll
      for (int j = 0; j < 4; ++j) {
        float s    = mag * reinterpret_cast<const float*>(&ws)[j];
        float mr   = reinterpret_cast<const float*>(&xr)[j] * s;
        float mi   = reinterpret_cast<const float*>(&xi)[j] * s;
        float base = reinterpret_cast<const float*>(&bo)[j] + tp;
        float wej  = reinterpret_cast<const float*>(&we)[j];
        float tor  = fmaf(mr, wej, base);
        float toi  = fmaf(mi, wej, base);
        float sor, cor, soi, coi;
        lut_sincos(sLut, tor, sor, cor);
        lut_sincos(sLut, toi, soi, coi);
        reinterpret_cast<float*>(&oR)[j] = cor * coi - sor * soi;
        reinterpret_cast<float*>(&oI)[j] = cor * soi + sor * coi;
      }
      po_r[d4] = oR;
      po_i[d4] = oI;
    }
  }
}

extern "C" void kernel_launch(void* const* d_in, const int* in_sizes, int n_in,
                              void* d_out, int out_size, void* d_ws, size_t ws_size,
                              hipStream_t stream) {
  (void)n_in; (void)d_ws; (void)ws_size; (void)out_size;
  const float* x_real = (const float*)d_in[0];
  const float* x_imag = (const float*)d_in[1];
  const float* t      = (const float*)d_in[2];
  const float* trr    = (const float*)d_in[3];
  const float* tri    = (const float*)d_in[4];
  const float* wq     = (const float*)d_in[5];
  const float* bq     = (const float*)d_in[6];
  const float* wo     = (const float*)d_in[7];
  const float* bo     = (const float*)d_in[8];
  const float* be     = (const float*)d_in[9];

  const int B  = in_sizes[2];       // t has B elements
  const int BD = in_sizes[0];       // B*D
  float* out_real = (float*)d_out;
  float* out_imag = (float*)d_out + (size_t)BD;

  const int blocks = (B + ROWS_PER_BLOCK - 1) / ROWS_PER_BLOCK;
  echo_fused<<<blocks, 256, 0, stream>>>(x_real, x_imag, t, trr, tri, wq, bq,
                                         wo, bo, be, out_real, out_imag, B);
}
